// EfficientAttention_8409545965831
// MI455X (gfx1250) — compile-verified, matrix operand order checked by reading, not run on hardware
//
#include <hip/hip_runtime.h>
#include <cstdint>

#define BDIM 4
#define NSEQ 8192
#define CDIM 384
#define HH   6
#define DD   64
#define ROWS 128   // rows per workgroup (8 waves x 16 rows)
#define LDT  136   // padded LDS row length (halves)

typedef __attribute__((ext_vector_type(16))) _Float16 v16h;
typedef __attribute__((ext_vector_type(8)))  float    v8f;

union FragH { v16h v; uint32_t u[8]; _Float16 h[16]; };
union FragC { v8f  v; float f[8]; };

__device__ __forceinline__ v8f wmma16(v16h a, v16h b, v8f c) {
  // D = A*B + C, f16 inputs, f32 accumulate (v_wmma_f32_16x16x32_f16)
  return __builtin_amdgcn_wmma_f32_16x16x32_f16(false, a, false, b, (short)0, c,
                                                false, false);
}

// Load a 16x32 f16 fragment (A layout; also B when memory is [N][K] row-major).
// Lane L: m = L&15 selects row; halves split K: lanes 0-15 K=0..7,16..23 etc.
__device__ __forceinline__ v16h ld_frag_h(const _Float16* p, int ld) {
  int lane = threadIdx.x & 31;
  int m = lane & 15, hlf = lane >> 4;
  FragH f;
  const uint32_t* row = (const uint32_t*)(p + m * ld);
  int kb = 4 * hlf;  // pair index base
#pragma unroll
  for (int j = 0; j < 4; ++j) {
    f.u[j]     = row[kb + j];       // K = 8*hlf + 2j, +1
    f.u[4 + j] = row[8 + kb + j];   // K = 16 + 8*hlf + 2j, +1
  }
  return f.v;
}

// Same fragment but sourced from fp32 memory, converted to f16 in-register.
__device__ __forceinline__ v16h ld_frag_f32(const float* p, int ld) {
  int lane = threadIdx.x & 31;
  int m = lane & 15, hlf = lane >> 4;
  FragH f;
  const float* row = p + m * ld;
  int kb = 8 * hlf;
#pragma unroll
  for (int j = 0; j < 4; ++j) {
    float2 x0 = *(const float2*)(row + kb + 2 * j);
    float2 x1 = *(const float2*)(row + 16 + kb + 2 * j);
    f.h[2 * j]     = (_Float16)x0.x;
    f.h[2 * j + 1] = (_Float16)x0.y;
    f.h[8 + 2 * j]     = (_Float16)x1.x;
    f.h[8 + 2 * j + 1] = (_Float16)x1.y;
  }
  return f.v;
}

// ---------------- prep: weight conversion + accumulator zeroing ----------------
__global__ __launch_bounds__(256) void prep_kernel(
    const float* __restrict__ Wq, const float* __restrict__ Wk,
    const float* __restrict__ Wv, const float* __restrict__ Wp,
    _Float16* __restrict__ wqT, _Float16* __restrict__ wkT,
    _Float16* __restrict__ wvT, _Float16* __restrict__ wpT,
    float* __restrict__ S, float* __restrict__ kvU) {
  int i = blockIdx.x * blockDim.x + threadIdx.x;
  if (i < HH * DD * DD) {
    int h = i / (DD * DD), r = i % (DD * DD);
    int d = r / DD, e = r % DD;
    int o = (h * DD + e) * DD + d;  // transpose to [h][e][d] for B-fragments
    wqT[o] = (_Float16)Wq[i];
    wkT[o] = (_Float16)Wk[i];
    wvT[o] = (_Float16)(Wv[i] * 0.125f);  // fold 1/sqrt(64)
  }
  if (i < CDIM * CDIM) wpT[i] = (_Float16)Wp[i];  // Wp[c][c'] already [N][K]
  if (i < BDIM * HH * DD) S[i] = 0.f;
  if (i < BDIM * HH * DD * DD) kvU[i] = 0.f;
}

// ---- qkv: projections + q feature-softmax + exp(k) colsums + partial kv ----
__global__ __launch_bounds__(256) void qkv_kernel(
    const float* __restrict__ x,
    const float* __restrict__ bq, const float* __restrict__ bk,
    const float* __restrict__ bv,
    const _Float16* __restrict__ wqT, const _Float16* __restrict__ wkT,
    const _Float16* __restrict__ wvT,
    _Float16* __restrict__ qs, float* __restrict__ S,
    float* __restrict__ kvU) {
  __shared__ _Float16 ekT[DD][LDT];  // [d][n_local]
  __shared__ _Float16 vT[DD][LDT];   // [e][n_local]
  __shared__ float sS[DD];

  const int nt = NSEQ / ROWS;
  int t  = blockIdx.x % nt;
  int bh = blockIdx.x / nt;
  int h  = bh % HH, b = bh / HH;
  int n0 = t * ROWS;
  int w    = threadIdx.x >> 5;
  int lane = threadIdx.x & 31;
  int col = lane & 15, hlf = lane >> 4;

  if (threadIdx.x < DD) sS[threadIdx.x] = 0.f;
  __syncthreads();

  // A fragments for this wave's 16-row strip of x (head slice, converted f32->f16)
  const float* xa = x + ((size_t)(b * NSEQ + n0 + 16 * w)) * CDIM + h * DD;
  v16h a0 = ld_frag_f32(xa, CDIM);       // k = 0..31
  v16h a1 = ld_frag_f32(xa + 32, CDIM);  // k = 32..63

  // ---- Q projection + rowwise softmax over 64 features ----
  FragC qa[4];
#pragma unroll
  for (int ct = 0; ct < 4; ++ct) {
    float bias = bq[h * DD + ct * 16 + col];
    FragC c;
#pragma unroll
    for (int r = 0; r < 8; ++r) c.f[r] = bias;
    const _Float16* wb = wqT + (size_t)(h * DD + ct * 16) * DD;
    c.v = wmma16(a0, ld_frag_h(wb, DD), c.v);
    c.v = wmma16(a1, ld_frag_h(wb + 32, DD), c.v);
    qa[ct] = c;
  }
#pragma unroll
  for (int r = 0; r < 8; ++r) {
    float m = qa[0].f[r];
#pragma unroll
    for (int ct = 1; ct < 4; ++ct) m = fmaxf(m, qa[ct].f[r]);
#pragma unroll
    for (int msk = 8; msk >= 1; msk >>= 1) m = fmaxf(m, __shfl_xor(m, msk, 16));
    float ev[4], sum = 0.f;
#pragma unroll
    for (int ct = 0; ct < 4; ++ct) { ev[ct] = __expf(qa[ct].f[r] - m); sum += ev[ct]; }
#pragma unroll
    for (int msk = 8; msk >= 1; msk >>= 1) sum += __shfl_xor(sum, msk, 16);
    float inv = 1.f / sum;
#pragma unroll
    for (int ct = 0; ct < 4; ++ct) qa[ct].f[r] = ev[ct] * inv;
  }
  {
    _Float16* qp = qs + ((size_t)bh * NSEQ + n0 + 16 * w) * DD;
#pragma unroll
    for (int ct = 0; ct < 4; ++ct)
#pragma unroll
      for (int r = 0; r < 8; ++r)
        qp[(size_t)(r + 8 * hlf) * DD + ct * 16 + col] = (_Float16)qa[ct].f[r];
  }

  // ---- K projection, exp (max-free: k ~ N(0,1)), column sums ----
#pragma unroll
  for (int ct = 0; ct < 4; ++ct) {
    float bias = bk[h * DD + ct * 16 + col];
    FragC c;
#pragma unroll
    for (int r = 0; r < 8; ++r) c.f[r] = bias;
    const _Float16* wb = wkT + (size_t)(h * DD + ct * 16) * DD;
    c.v = wmma16(a0, ld_frag_h(wb, DD), c.v);
    c.v = wmma16(a1, ld_frag_h(wb + 32, DD), c.v);
    float colsum = 0.f;
#pragma unroll
    for (int r = 0; r < 8; ++r) {
      float e = __expf(c.f[r]);
      colsum += e;
      ekT[ct * 16 + col][16 * w + r + 8 * hlf] = (_Float16)e;
    }
    colsum += __shfl_xor(colsum, 16, 32);  // combine row halves (same column)
    if (hlf == 0) atomicAdd(&sS[ct * 16 + col], colsum);
  }

  // ---- V projection (scale folded into weights/bias) ----
#pragma unroll
  for (int ct = 0; ct < 4; ++ct) {
    float bias = bv[h * DD + ct * 16 + col] * 0.125f;
    FragC c;
#pragma unroll
    for (int r = 0; r < 8; ++r) c.f[r] = bias;
    const _Float16* wb = wvT + (size_t)(h * DD + ct * 16) * DD;
    c.v = wmma16(a0, ld_frag_h(wb, DD), c.v);
    c.v = wmma16(a1, ld_frag_h(wb + 32, DD), c.v);
#pragma unroll
    for (int r = 0; r < 8; ++r)
      vT[ct * 16 + col][16 * w + r + 8 * hlf] = (_Float16)c.f[r];
  }

  __syncthreads();

  if (threadIdx.x < DD) atomicAdd(&S[bh * DD + threadIdx.x], sS[threadIdx.x]);

  // ---- partial kv[d,e] = sum_n ek[n,d] * v[n,e] over this 128-row tile ----
  float* kvb = kvU + (size_t)bh * DD * DD;
#pragma unroll
  for (int tt = 0; tt < 2; ++tt) {
    int tile = 2 * w + tt;
    int dt = tile >> 2, et = tile & 3;
    FragC c;
#pragma unroll
    for (int r = 0; r < 8; ++r) c.f[r] = 0.f;
#pragma unroll
    for (int ks = 0; ks < 4; ++ks) {
      v16h af = ld_frag_h(&ekT[dt * 16][ks * 32], LDT);
      v16h bf = ld_frag_h(&vT[et * 16][ks * 32], LDT);
      c.v = wmma16(af, bf, c.v);
    }
#pragma unroll
    for (int r = 0; r < 8; ++r)
      atomicAdd(&kvb[(dt * 16 + r + 8 * hlf) * DD + et * 16 + col], c.f[r]);
  }
}

// ---------------- finalize kv: normalize rows by S[d], store [e][d] f16 ----------------
__global__ __launch_bounds__(256) void kvfin_kernel(
    const float* __restrict__ kvU, const float* __restrict__ S,
    _Float16* __restrict__ kvT) {
  int i = blockIdx.x * blockDim.x + threadIdx.x;
  if (i >= BDIM * HH * DD * DD) return;
  int bh = i / (DD * DD), r = i % (DD * DD);
  int d = r / DD, e = r % DD;
  kvT[((size_t)bh * DD + e) * DD + d] = (_Float16)(kvU[i] / S[bh * DD + d]);
}

// ---------------- y = qs @ kv (per head) ----------------
__global__ __launch_bounds__(256) void av_kernel(
    const _Float16* __restrict__ qs, const _Float16* __restrict__ kvT,
    _Float16* __restrict__ y) {
  const int nt = NSEQ / ROWS;
  int t  = blockIdx.x % nt;
  int bh = blockIdx.x / nt;
  int h  = bh % HH, b = bh / HH;
  int n0 = t * ROWS;
  int w    = threadIdx.x >> 5;
  int lane = threadIdx.x & 31;
  int col = lane & 15, hlf = lane >> 4;

  const _Float16* qp = qs + ((size_t)bh * NSEQ + n0 + 16 * w) * DD;
  v16h a0 = ld_frag_h(qp, DD);
  v16h a1 = ld_frag_h(qp + 32, DD);
  _Float16* yp = y + ((size_t)(b * NSEQ + n0 + 16 * w)) * CDIM + h * DD;
#pragma unroll
  for (int ct = 0; ct < 4; ++ct) {
    const _Float16* kb = kvT + ((size_t)bh * DD + ct * 16) * DD;
    FragC c;
#pragma unroll
    for (int r = 0; r < 8; ++r) c.f[r] = 0.f;
    c.v = wmma16(a0, ld_frag_h(kb, DD), c.v);
    c.v = wmma16(a1, ld_frag_h(kb + 32, DD), c.v);
#pragma unroll
    for (int r = 0; r < 8; ++r)
      yp[(size_t)(r + 8 * hlf) * CDIM + ct * 16 + col] = (_Float16)c.f[r];
  }
}

// ---------------- out = y @ Wp^T + bp (K = 384, A-fragments hoisted) ----------------
__global__ __launch_bounds__(256) void proj_kernel(
    const _Float16* __restrict__ y, const _Float16* __restrict__ wpT,
    const float* __restrict__ bp, float* __restrict__ out) {
  const int nt = NSEQ / ROWS;
  int t = blockIdx.x % nt;
  int b = blockIdx.x / nt;
  int n0 = t * ROWS;
  int w    = threadIdx.x >> 5;
  int lane = threadIdx.x & 31;
  int col = lane & 15, hlf = lane >> 4;

  const _Float16* ya = y + ((size_t)(b * NSEQ + n0 + 16 * w)) * CDIM;
  v16h af[12];
#pragma unroll
  for (int ks = 0; ks < 12; ++ks) af[ks] = ld_frag_h(ya + ks * 32, CDIM);

  float* ob = out + ((size_t)(b * NSEQ + n0 + 16 * w)) * CDIM;
  for (int ct = 0; ct < 24; ++ct) {
    float bias = bp[ct * 16 + col];
    FragC c;
#pragma unroll
    for (int r = 0; r < 8; ++r) c.f[r] = bias;
    const _Float16* wb = wpT + (size_t)(ct * 16) * CDIM;
#pragma unroll
    for (int ks = 0; ks < 12; ++ks)
      c.v = wmma16(af[ks], ld_frag_h(wb + ks * 32, CDIM), c.v);
#pragma unroll
    for (int r = 0; r < 8; ++r)
      ob[(size_t)(r + 8 * hlf) * CDIM + ct * 16 + col] = c.f[r];
  }
}

extern "C" void kernel_launch(void* const* d_in, const int* in_sizes, int n_in,
                              void* d_out, int out_size, void* d_ws, size_t ws_size,
                              hipStream_t stream) {
  const float* x  = (const float*)d_in[0];
  const float* Wq = (const float*)d_in[1];
  const float* bq = (const float*)d_in[2];
  const float* Wk = (const float*)d_in[3];
  const float* bk = (const float*)d_in[4];
  const float* Wv = (const float*)d_in[5];
  const float* bv = (const float*)d_in[6];
  const float* Wp = (const float*)d_in[7];
  const float* bp = (const float*)d_in[8];
  float* out = (float*)d_out;

  char* ws = (char*)d_ws;
  size_t off = 0;
  auto alloc = [&](size_t bytes) -> void* {
    void* p = ws + off;
    off = (off + bytes + 255) & ~(size_t)255;
    return p;
  };
  _Float16* wqT = (_Float16*)alloc(sizeof(_Float16) * HH * DD * DD);
  _Float16* wkT = (_Float16*)alloc(sizeof(_Float16) * HH * DD * DD);
  _Float16* wvT = (_Float16*)alloc(sizeof(_Float16) * HH * DD * DD);
  _Float16* wpT = (_Float16*)alloc(sizeof(_Float16) * CDIM * CDIM);
  float*    S   = (float*)alloc(sizeof(float) * BDIM * HH * DD);
  float*    kvU = (float*)alloc(sizeof(float) * BDIM * HH * DD * DD);
  _Float16* kvT = (_Float16*)alloc(sizeof(_Float16) * BDIM * HH * DD * DD);
  _Float16* qs  = (_Float16*)alloc(sizeof(_Float16) * (size_t)BDIM * HH * NSEQ * DD);
  _Float16* y   = (_Float16*)alloc(sizeof(_Float16) * (size_t)BDIM * NSEQ * CDIM);

  prep_kernel<<<(CDIM * CDIM + 255) / 256, 256, 0, stream>>>(
      Wq, Wk, Wv, Wp, wqT, wkT, wvT, wpT, S, kvU);
  qkv_kernel<<<BDIM * HH * (NSEQ / ROWS), 256, 0, stream>>>(
      x, bq, bk, bv, wqT, wkT, wvT, qs, S, kvU);
  kvfin_kernel<<<(BDIM * HH * DD * DD + 255) / 256, 256, 0, stream>>>(kvU, S, kvT);
  av_kernel<<<BDIM * HH * (NSEQ / ROWS), 256, 0, stream>>>(qs, kvT, y);
  proj_kernel<<<BDIM * (NSEQ / ROWS), 256, 0, stream>>>(y, wpT, bp, out);

  (void)in_sizes; (void)n_in; (void)out_size; (void)ws_size;
}
